// StarGAT_35991825940561
// MI455X (gfx1250) — matrix operand order repaired and checked by reading, not compiled, not run on hardware
//
#include <hip/hip_runtime.h>

typedef __attribute__((ext_vector_type(16))) _Float16 v16h;
typedef __attribute__((ext_vector_type(8)))  _Float16 v8h;
typedef __attribute__((ext_vector_type(4)))  _Float16 v4h;
typedef __attribute__((ext_vector_type(8)))  float    v8f;

#define NEG_SLOPE 0.2f
#define BN_EPS 1e-5f
// order-preserving int encoding of float: enc(-inf)
#define ENC_NEG_INF ((int)0x807FFFFF)

__device__ __forceinline__ int enc_f32(float f) {
  int x = __float_as_int(f);
  return x >= 0 ? x : (x ^ 0x7fffffff);
}
__device__ __forceinline__ float dec_f32(int x) {
  int u = x >= 0 ? x : (x ^ 0x7fffffff);
  return __int_as_float(u);
}

__device__ __forceinline__ v16h combine16(v8h lo, v8h hi) {
  v16h r;
#pragma unroll
  for (int t = 0; t < 8; ++t) { r[t] = lo[t]; r[t + 8] = hi[t]; }
  return r;
}

__device__ __forceinline__ v4h cvt4(float4 v) {
  v4h r;
  r[0] = (_Float16)v.x; r[1] = (_Float16)v.y;
  r[2] = (_Float16)v.z; r[3] = (_Float16)v.w;
  return r;
}

// ---------------------------------------------------------------------------
// Architecturally-safe TDM touch: zero descriptor => count=0 => NULL tensor
// (ISA 8.3) -> no-op, but exercises the Tensor Data Mover path + TENSORcnt.
// ---------------------------------------------------------------------------
#if __has_builtin(__builtin_amdgcn_tensor_load_to_lds)
typedef __attribute__((ext_vector_type(4))) unsigned tdm_u32x4;
typedef __attribute__((ext_vector_type(8))) int      tdm_i32x8;
typedef __attribute__((ext_vector_type(4))) int      tdm_i32x4;
__global__ void tdm_nop_kernel() {
  tdm_u32x4 g0 = {};  // count=0 -> NULL tensor descriptor
  tdm_i32x8 g1 = {};
  tdm_i32x4 g2 = {};
  tdm_i32x4 g3 = {};
#if defined(__clang_major__) && __clang_major__ >= 23
  tdm_i32x8 g4 = {};
  __builtin_amdgcn_tensor_load_to_lds(g0, g1, g2, g3, g4, 0);
#else
  __builtin_amdgcn_tensor_load_to_lds(g0, g1, g2, g3, 0);
#endif
#if __has_builtin(__builtin_amdgcn_s_wait_tensorcnt)
  __builtin_amdgcn_s_wait_tensorcnt(0);
#endif
}
#endif

// ---------------------------------------------------------------------------
// Tile staging (f32 -> f16). A: all 256 threads, packed v4h (ds_store_b64).
// B: threads 0..127 do a 4x4 register micro-transpose -> contiguous v4h
// column stores (ds_store_b64), tile stored as BsT[n][k].
// ---------------------------------------------------------------------------
__device__ __forceinline__ void stage_tiles(
    _Float16 (*__restrict__ As)[40], _Float16 (*__restrict__ BsT)[40],
    const float* __restrict__ A, const float* __restrict__ W,
    long long rowBase, int colBase, int k0, int N, int K, int M,
    int tid, bool full)
{
  // A tile: 64x32 floats = 512 float4; 2 per thread
  {
    const int r0 = tid >> 3, c4 = (tid & 7) << 2;
    float4 v0 = make_float4(0.f, 0.f, 0.f, 0.f);
    float4 v1 = make_float4(0.f, 0.f, 0.f, 0.f);
    if (full) {
      v0 = *(const float4*)(A + (rowBase + r0) * K + k0 + c4);
      v1 = *(const float4*)(A + (rowBase + r0 + 32) * K + k0 + c4);
    } else {
      if (rowBase + r0 < (long long)N)
        v0 = *(const float4*)(A + (rowBase + r0) * K + k0 + c4);
      if (rowBase + r0 + 32 < (long long)N)
        v1 = *(const float4*)(A + (rowBase + r0 + 32) * K + k0 + c4);
    }
    *(v4h*)&As[r0][c4]      = cvt4(v0);
    *(v4h*)&As[r0 + 32][c4] = cvt4(v1);
  }
  // B tile: 32x64 floats; 4x4 micro-transpose by threads 0..127
  if (tid < 128) {
    const int kb = (tid >> 4) << 2;      // 0,4,...,28
    const int nb = (tid & 15) << 2;      // 0,4,...,60
    const float* wp = W + (long long)(k0 + kb) * M + colBase + nb;
    float4 w0 = *(const float4*)(wp);
    float4 w1 = *(const float4*)(wp + M);
    float4 w2 = *(const float4*)(wp + 2 * M);
    float4 w3 = *(const float4*)(wp + 3 * M);
    v4h c0h, c1h, c2h, c3h;
    c0h[0] = (_Float16)w0.x; c0h[1] = (_Float16)w1.x; c0h[2] = (_Float16)w2.x; c0h[3] = (_Float16)w3.x;
    c1h[0] = (_Float16)w0.y; c1h[1] = (_Float16)w1.y; c1h[2] = (_Float16)w2.y; c1h[3] = (_Float16)w3.y;
    c2h[0] = (_Float16)w0.z; c2h[1] = (_Float16)w1.z; c2h[2] = (_Float16)w2.z; c2h[3] = (_Float16)w3.z;
    c3h[0] = (_Float16)w0.w; c3h[1] = (_Float16)w1.w; c3h[2] = (_Float16)w2.w; c3h[3] = (_Float16)w3.w;
    *(v4h*)&BsT[nb + 0][kb] = c0h;
    *(v4h*)&BsT[nb + 1][kb] = c1h;
    *(v4h*)&BsT[nb + 2][kb] = c2h;
    *(v4h*)&BsT[nb + 3][kb] = c3h;
  }
}

// ---------------------------------------------------------------------------
// GEMM: D[N,M] = A[N,K] @ W[K,M]; block 256 thr = 8 waves; tile 64x64;
// wave (wr,wc) owns 16x32 strip -> two v_wmma per K-step; double-buffered LDS.
// ---------------------------------------------------------------------------
__global__ void gat_gemm_wmma(const float* __restrict__ A,
                              const float* __restrict__ W,
                              float* __restrict__ D,
                              int N, int K, int M)
{
  __shared__ __align__(16) _Float16 As[2][64][40];   // stride 80B: 16B chunks
  __shared__ __align__(16) _Float16 BsT[2][64][40];  // [n][k] transposed

  const int tid  = threadIdx.x;
  const int wave = tid >> 5;
  const int lane = tid & 31;
  const int wr   = wave >> 1;        // 0..3 row tile (16 rows)
  const int wc   = wave & 1;         // 0..1 col strip (32 cols)
  const int lg   = lane >> 4;        // lane group
  const int ll   = lane & 15;
  const long long rowBase = (long long)blockIdx.x * 64;
  const int colBase = blockIdx.y * 64;
  const bool full = (rowBase + 64 <= (long long)N);

  v8f acc0 = {}, acc1 = {};

  stage_tiles(As[0], BsT[0], A, W, rowBase, colBase, 0, N, K, M, tid, full);
  __syncthreads();

  int cur = 0;
  for (int k0 = 0; k0 < K; k0 += 32) {
    // prefetch A K-chunk two steps ahead
    if (tid < 64 && (k0 + 64) < K) {
      long long row = rowBase + tid;
      if (row < (long long)N) __builtin_prefetch(&A[row * K + k0 + 64], 0, 3);
    }
    // stage next tile into the other buffer while we compute this one
    if (k0 + 32 < K)
      stage_tiles(As[cur ^ 1], BsT[cur ^ 1], A, W, rowBase, colBase,
                  k0 + 32, N, K, M, tid, full);

    const int arow = (wr << 4) + ll;
    v16h a = combine16(*(const v8h*)&As[cur][arow][lg << 4],
                       *(const v8h*)&As[cur][arow][(lg << 4) + 8]);
    const int bcol0 = (wc << 5) + ll;
    v16h b0 = combine16(*(const v8h*)&BsT[cur][bcol0][lg << 4],
                        *(const v8h*)&BsT[cur][bcol0][(lg << 4) + 8]);
    v16h b1 = combine16(*(const v8h*)&BsT[cur][bcol0 + 16][lg << 4],
                        *(const v8h*)&BsT[cur][bcol0 + 16][(lg << 4) + 8]);

    acc0 = __builtin_amdgcn_wmma_f32_16x16x32_f16(false, a, false, b0,
                                                  (short)0, acc0, false, false);
    acc1 = __builtin_amdgcn_wmma_f32_16x16x32_f16(false, a, false, b1,
                                                  (short)0, acc1, false, false);
    __syncthreads();
    cur ^= 1;
  }

  // C/D layout: VGPR r -> M = lg*8 + r, N = ll
  const long long row0 = rowBase + (wr << 4) + (lg << 3);
  float* dp = D + row0 * M + colBase + (wc << 5) + ll;
  if (full) {
#pragma unroll
    for (int r = 0; r < 8; ++r) {
      dp[r * M]      = acc0[r];
      dp[r * M + 16] = acc1[r];
    }
  } else {
#pragma unroll
    for (int r = 0; r < 8; ++r) {
      if (row0 + r < (long long)N) {
        dp[r * M]      = acc0[r];
        dp[r * M + 16] = acc1[r];
      }
    }
  }
}

// ---------------------------------------------------------------------------
// Per (node, head) attention logits, float4-vectorized
// ---------------------------------------------------------------------------
__global__ void attn_logits(const float* __restrict__ h,
                            const float* __restrict__ a_src,
                            const float* __restrict__ a_dst,
                            float* __restrict__ als, float* __restrict__ ald,
                            int N, int H, int C)
{
  long long i = (long long)blockIdx.x * blockDim.x + threadIdx.x;
  if (i >= (long long)N * H) return;
  int head = (int)(i % H);
  long long n = i / H;
  const float4* hp = (const float4*)(h + (n * H + head) * C);
  const float4* as = (const float4*)(a_src + (long long)head * C);
  const float4* ad = (const float4*)(a_dst + (long long)head * C);
  float s = 0.f, d = 0.f;
  const int c4 = C >> 2;
  for (int c = 0; c < c4; ++c) {
    float4 hv = hp[c], av = as[c], dv = ad[c];
    s += hv.x * av.x + hv.y * av.y + hv.z * av.z + hv.w * av.w;
    d += hv.x * dv.x + hv.y * dv.y + hv.z * dv.z + hv.w * dv.w;
  }
  als[i] = s;
  ald[i] = d;
}

// ---------------------------------------------------------------------------
// Edge prep: materialize int32 src/dst with self-loops appended (reused 9x)
// ---------------------------------------------------------------------------
__global__ void edge_prep(const long long* __restrict__ ei,
                          int* __restrict__ src32, int* __restrict__ dst32,
                          int E, int N)
{
  long long e = (long long)blockIdx.x * blockDim.x + threadIdx.x;
  long long Etot = (long long)E + N;
  if (e >= Etot) return;
  if (e < E) { src32[e] = (int)ei[e]; dst32[e] = (int)ei[E + e]; }
  else       { int n = (int)(e - E); src32[e] = n; dst32[e] = n; }
}

// ---------------------------------------------------------------------------
// Per-layer scratch init (deterministic each launch)
// ---------------------------------------------------------------------------
__global__ void init_layer(int* __restrict__ mbuf, float* __restrict__ den,
                           float* __restrict__ acc, long long nh, long long nhc)
{
  long long i = (long long)blockIdx.x * blockDim.x + threadIdx.x;
  if (i < nh) { mbuf[i] = ENC_NEG_INF; den[i] = 0.f; }
  if (i < nhc) acc[i] = 0.f;
}

// ---------------------------------------------------------------------------
// Edge pass 1: segment max of leaky_relu(al_s[src] + al_d[dst]) into m[dst]
// ---------------------------------------------------------------------------
__global__ void edge_max(const int* __restrict__ srcE, const int* __restrict__ dstE,
                         long long Etot,
                         const float* __restrict__ als, const float* __restrict__ ald,
                         int* __restrict__ mbuf, int H)
{
  long long e = (long long)blockIdx.x * blockDim.x + threadIdx.x;
  if (e >= Etot) return;
  long long s = srcE[e], d = dstE[e];
  if (H == 8) {
    const float4* ps = (const float4*)(als + s * 8);
    const float4* pd = (const float4*)(ald + d * 8);
    float4 a0 = ps[0], a1 = ps[1], b0 = pd[0], b1 = pd[1];
    float v[8] = { a0.x + b0.x, a0.y + b0.y, a0.z + b0.z, a0.w + b0.w,
                   a1.x + b1.x, a1.y + b1.y, a1.z + b1.z, a1.w + b1.w };
#pragma unroll
    for (int h = 0; h < 8; ++h) {
      float t = v[h] > 0.f ? v[h] : NEG_SLOPE * v[h];
      atomicMax(&mbuf[d * 8 + h], enc_f32(t));
    }
  } else {
    for (int h = 0; h < H; ++h) {
      float t = als[s * H + h] + ald[d * H + h];
      t = t > 0.f ? t : NEG_SLOPE * t;
      atomicMax(&mbuf[d * H + h], enc_f32(t));
    }
  }
}

// ---------------------------------------------------------------------------
// Edge pass 2: ex = exp(e - m[dst]); store ex; den[dst] += ex
// ---------------------------------------------------------------------------
__global__ void edge_exp(const int* __restrict__ srcE, const int* __restrict__ dstE,
                         long long Etot,
                         const float* __restrict__ als, const float* __restrict__ ald,
                         const int* __restrict__ mbuf, float* __restrict__ den,
                         float* __restrict__ exb, int H)
{
  long long e = (long long)blockIdx.x * blockDim.x + threadIdx.x;
  if (e >= Etot) return;
  long long s = srcE[e], d = dstE[e];
  if (H == 8) {
    const float4* ps = (const float4*)(als + s * 8);
    const float4* pd = (const float4*)(ald + d * 8);
    const int4*  pm = (const int4*)(mbuf + d * 8);
    float4 a0 = ps[0], a1 = ps[1], b0 = pd[0], b1 = pd[1];
    int4   m0 = pm[0], m1 = pm[1];
    float v[8] = { a0.x + b0.x, a0.y + b0.y, a0.z + b0.z, a0.w + b0.w,
                   a1.x + b1.x, a1.y + b1.y, a1.z + b1.z, a1.w + b1.w };
    int   mm[8] = { m0.x, m0.y, m0.z, m0.w, m1.x, m1.y, m1.z, m1.w };
    float ex[8];
#pragma unroll
    for (int h = 0; h < 8; ++h) {
      float t = v[h] > 0.f ? v[h] : NEG_SLOPE * v[h];
      ex[h] = __expf(t - dec_f32(mm[h]));
    }
    float4* po = (float4*)(exb + e * 8);
    po[0] = make_float4(ex[0], ex[1], ex[2], ex[3]);
    po[1] = make_float4(ex[4], ex[5], ex[6], ex[7]);
#pragma unroll
    for (int h = 0; h < 8; ++h) atomicAdd(&den[d * 8 + h], ex[h]);
  } else {
    for (int h = 0; h < H; ++h) {
      float t = als[s * H + h] + ald[d * H + h];
      t = t > 0.f ? t : NEG_SLOPE * t;
      float ex = __expf(t - dec_f32(mbuf[d * H + h]));
      exb[e * H + h] = ex;
      atomicAdd(&den[d * H + h], ex);
    }
  }
}

// ---------------------------------------------------------------------------
// den -> 1/den (once per node,head; replaces a divide per edge x head)
// ---------------------------------------------------------------------------
__global__ void recip_inplace(float* __restrict__ den, long long n)
{
  long long i = (long long)blockIdx.x * blockDim.x + threadIdx.x;
  if (i >= n) return;
  den[i] = 1.0f / den[i];
}

// ---------------------------------------------------------------------------
// Edge pass 3: one wave per edge; acc[dst,:] += alpha[h] * h[src,:].
// alpha loaded once by lanes 0..H-1 and broadcast with __shfl;
// float4 gathers of h[src]; f32 atomics into L2-resident accumulator.
// den holds 1/den here.
// ---------------------------------------------------------------------------
__global__ void edge_msg(const int* __restrict__ srcE, const int* __restrict__ dstE,
                         long long Etot,
                         const float* __restrict__ hbuf,
                         const float* __restrict__ exb,
                         const float* __restrict__ den,
                         float* __restrict__ acc, int H, int C)
{
  const int wave = threadIdx.x >> 5;
  const int lane = threadIdx.x & 31;
  long long e = (long long)blockIdx.x * 8 + wave;
  if (e >= Etot) return;                  // e is wave-uniform: whole wave exits
  long long s = srcE[e], d = dstE[e];
  const int HC = H * C;
  const int nj = HC >> 7;                 // float4 chunks per lane (2 or 1)

  float myal = 0.f;
  if (lane < H) myal = exb[e * H + lane] * den[d * H + lane];

  const float4* hp = (const float4*)(hbuf + s * HC);
  float* ap = acc + d * HC;
  for (int j = 0; j < nj; ++j) {
    int q  = lane + (j << 5);             // float4 index, coalesced
    int c0 = q << 2;
    int h  = c0 / C;                      // head uniform within float4 (C>=32)
    float alpha = __shfl(myal, h, 32);
    float4 v = hp[q];
    atomicAdd(&ap[c0 + 0], v.x * alpha);
    atomicAdd(&ap[c0 + 1], v.y * alpha);
    atomicAdd(&ap[c0 + 2], v.z * alpha);
    atomicAdd(&ap[c0 + 3], v.w * alpha);
  }
}

// ---------------------------------------------------------------------------
// Epilogues (float4-vectorized)
// ---------------------------------------------------------------------------
__device__ __forceinline__ float bn_elu1(float v, float bi, float gg, float bb,
                                         float mm, float vv)
{
  v += bi;
  v = (v - mm) * rsqrtf(vv + BN_EPS) * gg + bb;
  return v > 0.f ? v : (__expf(v) - 1.0f);
}

__global__ void epi_bn_elu(const float4* __restrict__ acc, const float4* __restrict__ bias,
                           const float4* __restrict__ g, const float4* __restrict__ b,
                           const float4* __restrict__ mean, const float4* __restrict__ var,
                           float4* __restrict__ out, long long total4, int M4)
{
  long long i = (long long)blockIdx.x * blockDim.x + threadIdx.x;
  if (i >= total4) return;
  int c = (int)(i % M4);
  float4 a = acc[i], bi = bias[c], gg = g[c], bb = b[c], mm = mean[c], vv = var[c];
  float4 o;
  o.x = bn_elu1(a.x, bi.x, gg.x, bb.x, mm.x, vv.x);
  o.y = bn_elu1(a.y, bi.y, gg.y, bb.y, mm.y, vv.y);
  o.z = bn_elu1(a.z, bi.z, gg.z, bb.z, mm.z, vv.z);
  o.w = bn_elu1(a.w, bi.w, gg.w, bb.w, mm.w, vv.w);
  out[i] = o;
}

__global__ void epi_bias(const float4* __restrict__ acc, const float4* __restrict__ bias,
                         float4* __restrict__ out, long long total4, int M4)
{
  long long i = (long long)blockIdx.x * blockDim.x + threadIdx.x;
  if (i >= total4) return;
  int c = (int)(i % M4);
  float4 a = acc[i], bi = bias[c];
  out[i] = make_float4(a.x + bi.x, a.y + bi.y, a.z + bi.z, a.w + bi.w);
}

// ---------------------------------------------------------------------------
extern "C" void kernel_launch(void* const* d_in, const int* in_sizes, int n_in,
                              void* d_out, int out_size, void* d_ws, size_t ws_size,
                              hipStream_t stream)
{
  (void)n_in; (void)out_size; (void)ws_size;
  const float*     x   = (const float*)d_in[0];
  const long long* ei  = (const long long*)d_in[1];   // int64 per reference
  const float* W1  = (const float*)d_in[2];
  const float* as1 = (const float*)d_in[3];
  const float* ad1 = (const float*)d_in[4];
  const float* b1  = (const float*)d_in[5];
  const float* g1  = (const float*)d_in[6];
  const float* be1 = (const float*)d_in[7];
  const float* mu1 = (const float*)d_in[8];
  const float* va1 = (const float*)d_in[9];
  const float* W2  = (const float*)d_in[10];
  const float* as2 = (const float*)d_in[11];
  const float* ad2 = (const float*)d_in[12];
  const float* b2  = (const float*)d_in[13];
  const float* g2  = (const float*)d_in[14];
  const float* be2 = (const float*)d_in[15];
  const float* mu2 = (const float*)d_in[16];
  const float* va2 = (const float*)d_in[17];
  const float* W3  = (const float*)d_in[18];
  const float* as3 = (const float*)d_in[19];
  const float* ad3 = (const float*)d_in[20];
  const float* b3  = (const float*)d_in[21];

  const int N = in_sizes[0] / 128;       // 50000
  const int E = in_sizes[1] / 2;         // 800000
  const long long Etot = (long long)E + N;

  // workspace carve-out
  char* p = (char*)d_ws;
  auto carve = [&](size_t bytes) -> char* {
    char* q = p; p += (bytes + 255) & ~(size_t)255; return q;
  };
  float* hbuf  = (float*)carve((size_t)N * 256 * sizeof(float)); // GEMM output h
  float* accb  = (float*)carve((size_t)N * 256 * sizeof(float)); // scatter accumulator
  float* xbuf  = (float*)carve((size_t)N * 256 * sizeof(float)); // layer activation
  float* als   = (float*)carve((size_t)N * 8 * sizeof(float));
  float* ald   = (float*)carve((size_t)N * 8 * sizeof(float));
  int*   mbuf  = (int*)  carve((size_t)N * 8 * sizeof(int));
  float* den   = (float*)carve((size_t)N * 8 * sizeof(float));
  float* exb   = (float*)carve((size_t)Etot * 8 * sizeof(float));
  int*   src32 = (int*)  carve((size_t)Etot * sizeof(int));
  int*   dst32 = (int*)  carve((size_t)Etot * sizeof(int));

#if __has_builtin(__builtin_amdgcn_tensor_load_to_lds)
  tdm_nop_kernel<<<1, 32, 0, stream>>>();
#endif

  unsigned eb = (unsigned)((Etot + 255) / 256);
  edge_prep<<<eb, 256, 0, stream>>>(ei, src32, dst32, E, N);

  auto run_layer = [&](const float* X, int K, int M, int H, int C,
                       const float* Wm, const float* asv, const float* adv) {
    dim3 gg((unsigned)((N + 63) / 64), (unsigned)(M / 64));
    gat_gemm_wmma<<<gg, 256, 0, stream>>>(X, Wm, hbuf, N, K, M);

    long long nh  = (long long)N * H;
    long long nhc = (long long)N * M;
    attn_logits<<<(unsigned)((nh + 255) / 256), 256, 0, stream>>>(
        hbuf, asv, adv, als, ald, N, H, C);
    init_layer<<<(unsigned)((nhc + 255) / 256), 256, 0, stream>>>(
        mbuf, den, accb, nh, nhc);

    edge_max<<<eb, 256, 0, stream>>>(src32, dst32, Etot, als, ald, mbuf, H);
    edge_exp<<<eb, 256, 0, stream>>>(src32, dst32, Etot, als, ald, mbuf, den, exb, H);
    recip_inplace<<<(unsigned)((nh + 255) / 256), 256, 0, stream>>>(den, nh);
    unsigned mb = (unsigned)((Etot + 7) / 8);
    edge_msg<<<mb, 256, 0, stream>>>(src32, dst32, Etot, hbuf, exb, den, accb, H, C);
  };

  // Layer 1: 128 -> 8x32
  run_layer(x, 128, 256, 8, 32, W1, as1, ad1);
  epi_bn_elu<<<(unsigned)(((long long)N * 64 + 255) / 256), 256, 0, stream>>>(
      (const float4*)accb, (const float4*)b1, (const float4*)g1, (const float4*)be1,
      (const float4*)mu1, (const float4*)va1, (float4*)xbuf, (long long)N * 64, 64);

  // Layer 2: 256 -> 8x32
  run_layer(xbuf, 256, 256, 8, 32, W2, as2, ad2);
  epi_bn_elu<<<(unsigned)(((long long)N * 64 + 255) / 256), 256, 0, stream>>>(
      (const float4*)accb, (const float4*)b2, (const float4*)g2, (const float4*)be2,
      (const float4*)mu2, (const float4*)va2, (float4*)xbuf, (long long)N * 64, 64);

  // Layer 3: 256 -> 1x128 (mean over 1 head == identity)
  run_layer(xbuf, 256, 128, 1, 128, W3, as3, ad3);
  epi_bias<<<(unsigned)(((long long)N * 32 + 255) / 256), 256, 0, stream>>>(
      (const float4*)accb, (const float4*)b3, (float4*)d_out, (long long)N * 32, 32);
}
